// triplet_loss_cl_1709396984281
// MI455X (gfx1250) — matrix operand order repaired and checked by reading, not compiled, not run on hardware
//
#include <hip/hip_runtime.h>
#include <math.h>

// ---- types ----
typedef __attribute__((ext_vector_type(16))) _Float16 v16h;
typedef __attribute__((ext_vector_type(8)))  _Float16 v8h;
typedef __attribute__((ext_vector_type(8)))  float    v8f;

#define N_ROWS       8192
#define DIMK         256
#define WAVES_PER_WG 4
#define THREADS      (WAVES_PER_WG * 32)
#define ROWS_PER_WG  (WAVES_PER_WG * 16)   // 64
#define LDS_PITCH    264                    // 256 + 8 halfword pad -> conflict-free b128 reads
#define SHIFT_C      40.0f
#define EPS_LOSS     1e-5f

__global__ __launch_bounds__(THREADS)
void diag_softmax_loss_kernel(const float* __restrict__ q,
                              const float* __restrict__ g,
                              float* __restrict__ partials)
{
    // double-buffered G tile: [buf][16 rows][padded K] f16 hi/lo planes (~34 KB)
    __shared__ _Float16 sHi[2][16 * LDS_PITCH];
    __shared__ _Float16 sLo[2][16 * LDS_PITCH];

    const int tid  = threadIdx.x;
    const int wave = tid >> 5;
    const int lane = tid & 31;
    const int rowBase = blockIdx.x * ROWS_PER_WG + wave * 16;

    // ------------------------------------------------------------------
    // A-fragments: this wave's 16xK Q strip as f16 hi/lo, kept in VGPRs.
    // A 16x32 layout: lane m=lane&15; lanes<16 hold K {0..7,16..23},
    // lanes>=16 hold K {8..15,24..31} within each K=32 chunk.
    // ------------------------------------------------------------------
    const int m   = lane & 15;
    const int kbA = (lane < 16) ? 0 : 8;
    const float* qrow = q + (size_t)(rowBase + m) * DIMK;

    v16h Ahi[8], Alo[8];
#pragma unroll
    for (int c = 0; c < 8; ++c) {
        v16h uh, ul;
#pragma unroll
        for (int e = 0; e < 16; ++e) {
            int k = c * 32 + kbA + e + ((e < 8) ? 0 : 8);
            float x = qrow[k];
            _Float16 h = (_Float16)x;           // v_cvt_f16_f32 (RNE)
            uh[e] = h;
            ul[e] = (_Float16)(x - (float)h);   // residual, ~2^-11 of x
        }
        Ahi[c] = uh;
        Alo[c] = ul;
    }

    // B 32x16 layout: lane holds column n=lane&15; lanes<16 hold K {0..7,16..23},
    // lanes>=16 hold K {8..15,24..31} of each chunk (same order as A).
    const int kbB = (lane < 16) ? 0 : 8;

    float s[8];
#pragma unroll
    for (int r = 0; r < 8; ++r) s[r] = 0.0f;
    float diag = 0.0f;

    const int diagTile = rowBase >> 4;
    const int NT = N_ROWS / 16;                 // 512 column tiles

    // staging coordinates: 4096 elements / 128 threads = 32 contiguous each
    const int grow = tid >> 3;
    const int gcol = (tid & 7) * 32;

    // ---- prologue: stage tile 0 into buffer 0 ----
    {
        const float* gp = g + (size_t)grow * DIMK + gcol;
        _Float16* hp = &sHi[0][grow * LDS_PITCH + gcol];
        _Float16* lp = &sLo[0][grow * LDS_PITCH + gcol];
#pragma unroll
        for (int v = 0; v < 4; ++v) {
            v8h vh, vl;
#pragma unroll
            for (int e = 0; e < 8; ++e) {
                float x = gp[v * 8 + e];
                _Float16 h = (_Float16)x;
                vh[e] = h;
                vl[e] = (_Float16)(x - (float)h);
            }
            *(v8h*)(hp + v * 8) = vh;
            *(v8h*)(lp + v * 8) = vl;
        }
    }

    for (int j = 0; j < NT; ++j) {
        __syncthreads();   // buf[j&1] staged; buf[(j+1)&1] free to overwrite
        const int cur = j & 1;

        // ---- stage tile j+1 into the other buffer (overlaps with WMMAs) ----
        if (j + 1 < NT) {
            const int nxt = (j + 1) & 1;
            const float* gp = g + (size_t)((j + 1) * 16 + grow) * DIMK + gcol;
            _Float16* hp = &sHi[nxt][grow * LDS_PITCH + gcol];
            _Float16* lp = &sLo[nxt][grow * LDS_PITCH + gcol];
#pragma unroll
            for (int v = 0; v < 4; ++v) {
                v8h vh, vl;
#pragma unroll
                for (int e = 0; e < 8; ++e) {
                    float x = gp[v * 8 + e];
                    _Float16 h = (_Float16)x;
                    vh[e] = h;
                    vl[e] = (_Float16)(x - (float)h);
                }
                *(v8h*)(hp + v * 8) = vh;
                *(v8h*)(lp + v * 8) = vl;
            }
            if (j + 2 < NT) __builtin_prefetch(gp + 16 * DIMK, 0, 3);
        }

        // ---- 16x16 logits tile, 3 independent WMMA chains ----
        v8f aHH = {}, aHL = {}, aLH = {};
#pragma unroll
        for (int c = 0; c < 8; ++c) {
            const v8h* ph = (const v8h*)&sHi[cur][m * LDS_PITCH + c * 32 + kbB];
            const v8h* pl = (const v8h*)&sLo[cur][m * LDS_PITCH + c * 32 + kbB];
            v16h Bhi = __builtin_shufflevector(ph[0], ph[2],
                         0,1,2,3,4,5,6,7,8,9,10,11,12,13,14,15);
            v16h Blo = __builtin_shufflevector(pl[0], pl[2],
                         0,1,2,3,4,5,6,7,8,9,10,11,12,13,14,15);
            aHH = __builtin_amdgcn_wmma_f32_16x16x32_f16(false, Ahi[c], false, Bhi,
                                                         (short)0, aHH, false, false);
            aHL = __builtin_amdgcn_wmma_f32_16x16x32_f16(false, Ahi[c], false, Blo,
                                                         (short)0, aHL, false, false);
            aLH = __builtin_amdgcn_wmma_f32_16x16x32_f16(false, Alo[c], false, Bhi,
                                                         (short)0, aLH, false, false);
        }
        v8f acc = aHH + (aHL + aLH);

        // capture diagonal logit when this column tile crosses our rows
        if (j == diagTile) {
            int want = (lane < 8) ? lane : ((lane >= 24) ? (lane - 24) : -1);
#pragma unroll
            for (int r = 0; r < 8; ++r)
                if (r == want) diag = acc[r];
        }

        // streaming softmax denominator, fixed shift (logits ~ N(0,256): no overflow)
#pragma unroll
        for (int r = 0; r < 8; ++r)
            s[r] += __expf(acc[r] - SHIFT_C);
    }

    // ---- reduce row sums across the 16 lanes holding each row ----
#pragma unroll
    for (int r = 0; r < 8; ++r) {
        float v = s[r];
        v += __shfl_xor(v, 1, 16);
        v += __shfl_xor(v, 2, 16);
        v += __shfl_xor(v, 4, 16);
        v += __shfl_xor(v, 8, 16);
        s[r] = v;
    }

    // ---- per-row loss contribution; sum over the wave ----
    float contrib = 0.0f;
    {
        int want = (lane < 8) ? lane : ((lane >= 24) ? (lane - 24) : -1);
        if (want >= 0) {
            float srow = 1.0f;
#pragma unroll
            for (int r = 0; r < 8; ++r)
                if (r == want) srow = s[r];
            float p = __expf(diag - SHIFT_C) / srow;
            contrib = -__logf(p + EPS_LOSS) * (1.0f / (float)N_ROWS);
        }
    }
    contrib += __shfl_xor(contrib, 1, 32);
    contrib += __shfl_xor(contrib, 2, 32);
    contrib += __shfl_xor(contrib, 4, 32);
    contrib += __shfl_xor(contrib, 8, 32);
    contrib += __shfl_xor(contrib, 16, 32);
    if (lane == 0)
        partials[blockIdx.x * WAVES_PER_WG + wave] = contrib;
}

__global__ __launch_bounds__(512)
void reduce_partials_kernel(const float* __restrict__ partials,
                            float* __restrict__ out, int n)
{
    __shared__ float buf[512];
    int t = threadIdx.x;
    buf[t] = (t < n) ? partials[t] : 0.0f;
    __syncthreads();
#pragma unroll
    for (int st = 256; st > 0; st >>= 1) {
        if (t < st) buf[t] += buf[t + st];
        __syncthreads();
    }
    if (t == 0) out[0] = buf[0];
}

extern "C" void kernel_launch(void* const* d_in, const int* in_sizes, int n_in,
                              void* d_out, int out_size, void* d_ws, size_t ws_size,
                              hipStream_t stream) {
    const float* q = (const float*)d_in[0];
    const float* g = (const float*)d_in[1];
    float* partials = (float*)d_ws;

    const int nBlocks = N_ROWS / ROWS_PER_WG;                 // 128 workgroups
    diag_softmax_loss_kernel<<<dim3(nBlocks), dim3(THREADS), 0, stream>>>(q, g, partials);
    reduce_partials_kernel<<<dim3(1), dim3(512), 0, stream>>>(partials, (float*)d_out,
                                                              nBlocks * WAVES_PER_WG);
}
